// CustomDense_8770323218906
// MI455X (gfx1250) — 8-bit matrix operand order repaired and checked by reading, compiled, not run on hardware
//
#include <hip/hip_runtime.h>


typedef __attribute__((ext_vector_type(8))) int v8i;
typedef __attribute__((ext_vector_type(4))) int v4i;
typedef __attribute__((ext_vector_type(2))) int v2i;

#define KDIM   16384
#define BDIM   64
#define UDIM   1024
#define DDIM   1024

__global__ __launch_bounds__(256) void build_ahat(
    const float* __restrict__ x, unsigned char* __restrict__ Ahat) {
  int idx = blockIdx.x * blockDim.x + threadIdx.x;
  int v = (int)x[idx];
  unsigned int one = 1u << ((v & 3) * 8);
  uint4 oh;
  oh.x = ((v >> 2) == 0) ? one : 0u;
  oh.y = ((v >> 2) == 1) ? one : 0u;
  oh.z = ((v >> 2) == 2) ? one : 0u;
  oh.w = ((v >> 2) == 3) ? one : 0u;
  ((uint4*)Ahat)[idx] = oh;
}

__global__ __launch_bounds__(256) void build_ghat(
    const float* __restrict__ w, signed char* __restrict__ GhatT) {
  int idx = blockIdx.x * blockDim.x + threadIdx.x;
  int u = idx >> 10;
  int d = idx & (DDIM - 1);
  int iw = (int)w[d * UDIM + u];
  unsigned int wd[4];
#pragma unroll
  for (int j = 0; j < 4; ++j) {
    unsigned int pack = 0;
#pragma unroll
    for (int b = 0; b < 4; ++b) {
      int v = 4 * j + b;
      int q = (v * iw) / 256;
      pack |= ((unsigned int)(q & 0xff)) << (8 * b);
    }
    wd[j] = pack;
  }
  uint4 st = make_uint4(wd[0], wd[1], wd[2], wd[3]);
  *(uint4*)(GhatT + (size_t)u * KDIM + (size_t)d * 16) = st;
}

__global__ __launch_bounds__(256) void bitserial_wmma_gemm(
    const unsigned char* __restrict__ Ahat,
    const signed char*  __restrict__ GhatT,
    const float* __restrict__ bias,
    float* __restrict__ out) {
  const int lane  = threadIdx.x & 31;
  const int wave  = blockIdx.x * 8 + (threadIdx.x >> 5);
  const int mtile = wave >> 6;
  const int ntile = wave & 63;
  const int m     = lane & 15;
  const int half  = lane >> 4;

  const unsigned char* __restrict__ arow =
      Ahat + (size_t)(mtile * 16 + m) * KDIM;
  const signed char* __restrict__ brow =
      GhatT + (size_t)(ntile * 16 + m) * KDIM;

  union FragA { v8i v; v2i h[4]; };
  union FragB { v8i v; v4i q[2]; };

  v8i acc0 = {};
  v8i acc1 = {};

  for (int k0 = 0; k0 < KDIM; k0 += 128) {
    __builtin_prefetch(brow + k0 + 1024, 0, 1);

    FragA a0, a1; FragB b0, b1;
#pragma unroll
    for (int j = 0; j < 4; ++j) {
      a0.h[j] = *(const v2i*)(arow + k0 +       32 * (j >> 1) + 8 * (j & 1) + 16 * half);
      a1.h[j] = *(const v2i*)(arow + k0 + 64 +  32 * (j >> 1) + 8 * (j & 1) + 16 * half);
    }
    b0.q[0] = *(const v4i*)(brow + k0 +       16 * half);
    b0.q[1] = *(const v4i*)(brow + k0 + 32 +  16 * half);
    b1.q[0] = *(const v4i*)(brow + k0 + 64 +  16 * half);
    b1.q[1] = *(const v4i*)(brow + k0 + 96 +  16 * half);

    acc0 = __builtin_amdgcn_wmma_i32_16x16x64_iu8(
         false, a0.v,  true, b0.v, acc0,
         false,  false);
    acc1 = __builtin_amdgcn_wmma_i32_16x16x64_iu8(
         false, a1.v,  true, b1.v, acc1,
         false,  false);
  }

  const int n = ntile * 16 + m;
  const float bv = bias[n];
#pragma unroll
  for (int r = 0; r < 8; ++r) {
    float f = (float)(acc0[r] + acc1[r]) + bv;
    int row = mtile * 16 + r + 8 * half;
    out[(size_t)row * UDIM + n] = f > 0.0f ? f : 0.0f;
  }
}

extern "C" void kernel_launch(void* const* d_in, const int* in_sizes, int n_in,
                              void* d_out, int out_size, void* d_ws, size_t ws_size,
                              hipStream_t stream) {
  const float* x    = (const float*)d_in[0];
  const float* w    = (const float*)d_in[2];
  const float* bias = (const float*)d_in[3];
  float* out        = (float*)d_out;

  unsigned char* Ahat  = (unsigned char*)d_ws;
  signed char*   GhatT = (signed char*)d_ws + (1u << 20);

  build_ahat<<<(BDIM * DDIM) / 256, 256, 0, stream>>>(x, Ahat);
  build_ghat<<<(DDIM * UDIM) / 256, 256, 0, stream>>>(w, GhatT);
  bitserial_wmma_gemm<<<32, 256, 0, stream>>>(Ahat, GhatT, bias, out);
}
